// Transformer_47648367182337
// MI455X (gfx1250) — matrix operand order repaired and checked by reading, not compiled, not run on hardware
//
#include <hip/hip_runtime.h>
#include <hip/hip_bf16.h>

#define DEV __device__ __forceinline__

typedef unsigned short u16;
typedef __attribute__((ext_vector_type(16))) __bf16 bf16x16;
typedef __attribute__((ext_vector_type(8)))  __bf16 bf16x8;
typedef __attribute__((ext_vector_type(8)))  float  f32x8;
typedef __attribute__((ext_vector_type(4)))  unsigned int u32x4;
typedef __attribute__((ext_vector_type(8)))  int i32x8;
typedef __attribute__((ext_vector_type(4)))  int i32x4;

union FragBF { bf16x16 v; bf16x8 h[2]; };

static constexpr int Dm  = 1024;   // d_model
static constexpr int DFFc = 4096;  // d_ff
static constexpr int Hh  = 16;     // heads
static constexpr int Lc  = 6;      // layers
static constexpr int DHc = 64;     // head dim
static constexpr int Bc  = 8;      // batch
static constexpr int Sc  = 1024;   // seq
static constexpr int Mrows = Bc * Sc;  // 8192 token rows

#if defined(__has_builtin)
# if __has_builtin(__builtin_amdgcn_tensor_load_to_lds)
#  define HAVE_TDM 1
# endif
#endif
#ifndef HAVE_TDM
# define HAVE_TDM 0
#endif

DEV u16 f2bf(float f) {
  union { float f; unsigned u; } x; x.f = f;
  unsigned r = x.u + 0x7fffu + ((x.u >> 16) & 1u);   // round-to-nearest-even
  return (u16)(r >> 16);
}
DEV unsigned pack_bf2(float a, float b) {
  return (unsigned)f2bf(a) | ((unsigned)f2bf(b) << 16);
}

DEV f32x8 wmma_bf16(bf16x16 a, bf16x16 b, f32x8 c) {
#if defined(__HIP_DEVICE_COMPILE__)
  // D = A(16x32 bf16) * B(32x16 bf16) + C(16x16 f32)
  return __builtin_amdgcn_wmma_f32_16x16x32_bf16(false, a, false, b, (short)0, c, false, false);
#else
  return c;
#endif
}

// ---------------------------------------------------------------------------
// Tensor Data Mover: 2D bf16 tile (tRows x tCols elems, row stride
// strideElems) from global -> LDS at byte offset ldsOff, with LDS row padding
// of (amountCode+1) dwords every 2^(intervalCode+1) dwords (ISA 08 §8.3-8.4).
// Issue from ONE wave; follow with s_wait_tensorcnt 0 + block barrier.
// ---------------------------------------------------------------------------
DEV void tdm_load_2d(const u16* gsrc, unsigned ldsOff, unsigned strideElems,
                     unsigned tRows, unsigned tCols, int intervalCode, int amountCode)
{
#if defined(__HIP_DEVICE_COMPILE__) && HAVE_TDM
  unsigned long long ga = (unsigned long long)(size_t)gsrc;
  u32x4 g0;
  g0[0] = 1u;                                            // count=1, user desc
  g0[1] = ldsOff;                                        // lds_addr
  g0[2] = (unsigned)ga;                                  // global_addr[31:0]
  g0[3] = (unsigned)((ga >> 32) & 0x01ffffffull) | (2u << 30);  // [56:32]|type=2
  i32x8 g1;
  g1[0] = (1 << 16)                 // data_size = 2B
        | (1 << 20)                 // pad_enable
        | (intervalCode << 22)      // pad every 2^(c+1) dwords
        | (amountCode << 25);       // pad amount = c+1 dwords
  g1[1] = (int)((strideElems & 0xffffu) << 16);                          // tensor_dim0 lo
  g1[2] = (int)((strideElems >> 16) & 0xffffu) | (int)((tRows & 0xffffu) << 16); // dim0 hi | dim1 lo
  g1[3] = (int)((tRows >> 16) & 0xffffu) | (int)(tCols << 16);           // dim1 hi | tile_dim0
  g1[4] = (int)(tRows & 0xffffu);                                        // tile_dim1 (tile_dim2=0)
  g1[5] = (int)strideElems;                                              // tensor_dim0_stride lo32
  g1[6] = 0;
  g1[7] = 0;
  i32x4 z4 = {0, 0, 0, 0};
#if __clang_major__ >= 23
  i32x8 z8 = {0, 0, 0, 0, 0, 0, 0, 0};
  __builtin_amdgcn_tensor_load_to_lds(g0, g1, z4, z4, z8, 0);
#else
  __builtin_amdgcn_tensor_load_to_lds(g0, g1, z4, z4, 0);
#endif
#else
  (void)gsrc; (void)ldsOff; (void)strideElems; (void)tRows; (void)tCols;
  (void)intervalCode; (void)amountCode;
#endif
}

DEV void wait_tensor0() {
#if defined(__HIP_DEVICE_COMPILE__) && HAVE_TDM
  __builtin_amdgcn_s_wait_tensorcnt(0);
#endif
}

// ---------------------------------------------------------------------------
// Weight prep: f32 [K,N] (batched) -> bf16 [N,K]
// ---------------------------------------------------------------------------
__global__ __launch_bounds__(256) void transpose_bf16_kernel(
    const float* __restrict__ in, u16* __restrict__ out,
    int K, int N, long long inStride, long long outStride)
{
  long long z = blockIdx.z;
  int idx = blockIdx.x * 256 + threadIdx.x;
  if (idx >= K * N) return;
  int n = idx % N, kk = idx / N;
  out[z * outStride + (size_t)n * K + kk] = f2bf(in[z * inStride + idx]);
}

// ---------------------------------------------------------------------------
// Embedding * sqrt(D) + sinusoidal positional encoding
// ---------------------------------------------------------------------------
__global__ __launch_bounds__(256) void embed_kernel(
    const int* __restrict__ tokens, const float* __restrict__ emb,
    float* __restrict__ x)
{
  int idx = blockIdx.x * 256 + threadIdx.x;       // over Mrows*Dm
  int d  = idx & (Dm - 1);
  int bs = idx >> 10;                              // b*S + s
  int s  = bs & (Sc - 1);
  int tok = tokens[bs];
  float val = emb[(size_t)tok * Dm + d] * 32.0f;   // sqrt(1024)
  float freq = __expf(-9.210340371976184f * (float)(d & ~1) * (1.0f / Dm));
  float ang = (float)s * freq;
  val += (d & 1) ? cosf(ang) : sinf(ang);
  x[idx] = val;
}

// ---------------------------------------------------------------------------
// Generic WMMA GEMM:  C[M,N] = epilogue(A[M,K] @ Bw^T)   (Bw is bf16 [N,K])
// BM=BN=128, BK=32; 256 threads = 8 waves; wave grid 2x4, each wave 64x32.
// B tile (and bf16 A tile) fetched by the Tensor Data Mover into padded LDS.
// ---------------------------------------------------------------------------
template<typename TA, typename TC, bool BIAS, bool RES, bool RELU>
__global__ __launch_bounds__(256) void gemm_bf16_kernel(
    const TA* __restrict__ A, const u16* __restrict__ Bw,
    const float* __restrict__ bias, const float* __restrict__ res,
    TC* __restrict__ C, int M, int N, int K)
{
  constexpr int BM = 128, BN = 128, BK = 32, LDT = 40;  // pad 8 -> 80B row pitch
  __shared__ u16 sA[BM * LDT];
  __shared__ u16 sB[BN * LDT];
  const int tid  = threadIdx.x;
  const int lane = tid & 31, wave = tid >> 5;
  const int lh = lane >> 4, lidx = lane & 15;
  const int wm = wave & 1, wn = wave >> 1;
  const int m0 = blockIdx.y * BM, n0 = blockIdx.x * BN;
  const unsigned sAoff = (unsigned)(size_t)(void*)sA;
  const unsigned sBoff = (unsigned)(size_t)(void*)sB;

  f32x8 acc[4][2] = {};

  for (int k0 = 0; k0 < K; k0 += BK) {
    if constexpr (sizeof(TA) == 4) {            // f32 A -> bf16 in LDS (packed b32)
#pragma unroll
      for (int i = 0; i < 4; i++) {
        int chunk = tid + i * 256;               // 1024 float4 chunks
        int r = chunk >> 3, c4 = (chunk & 7) * 4;
        float4 vv = *(const float4*)((const float*)A + (size_t)(m0 + r) * K + k0 + c4);
        unsigned* dst = (unsigned*)&sA[r * LDT + c4];
        dst[0] = pack_bf2(vv.x, vv.y);
        dst[1] = pack_bf2(vv.z, vv.w);
      }
    }
#if HAVE_TDM
    if (tid < 32) {                              // wave 0 drives the TDM
      if constexpr (sizeof(TA) == 2)
        tdm_load_2d((const u16*)A + (size_t)m0 * K + k0, sAoff, K, BM, BK, 3, 3);
      tdm_load_2d(Bw + (size_t)n0 * K + k0, sBoff, K, BN, BK, 3, 3);
      wait_tensor0();
    }
#else
    if constexpr (sizeof(TA) == 2) {
#pragma unroll
      for (int i = 0; i < 2; i++) {
        int chunk = tid + i * 256;
        int r = chunk >> 2, c8 = (chunk & 3) * 8;
        *(uint4*)&sA[r * LDT + c8] =
            *(const uint4*)((const u16*)A + (size_t)(m0 + r) * K + k0 + c8);
      }
    }
#pragma unroll
    for (int i = 0; i < 2; i++) {
      int chunk = tid + i * 256;
      int r = chunk >> 2, c8 = (chunk & 3) * 8;
      *(uint4*)&sB[r * LDT + c8] =
          *(const uint4*)(Bw + (size_t)(n0 + r) * K + k0 + c8);
    }
#endif
    __syncthreads();

    const int kb = lh * 8, kb2 = lh * 16;
    FragBF af[4], bfg[2];
#pragma unroll
    for (int mt = 0; mt < 4; mt++) {             // A layout: M=lidx, K split
      int m = wm * 64 + mt * 16 + lidx;
      af[mt].h[0] = *(const bf16x8*)&sA[m * LDT + kb];
      af[mt].h[1] = *(const bf16x8*)&sA[m * LDT + kb + 16];
    }
#pragma unroll
    for (int nt = 0; nt < 2; nt++) {             // B layout: N=lidx, K split
      int n = wn * 32 + nt * 16 + lidx;
      bfg[nt].h[0] = *(const bf16x8*)&sB[n * LDT + kb];
      bfg[nt].h[1] = *(const bf16x8*)&sB[n * LDT + kb + 16];
    }
#pragma unroll
    for (int mt = 0; mt < 4; mt++)
#pragma unroll
      for (int nt = 0; nt < 2; nt++)
        acc[mt][nt] = wmma_bf16(af[mt].v, bfg[nt].v, acc[mt][nt]);
    __syncthreads();
  }

#pragma unroll
  for (int mt = 0; mt < 4; mt++)
#pragma unroll
    for (int nt = 0; nt < 2; nt++) {
      int col = n0 + wn * 32 + nt * 16 + lidx;
      float bval = BIAS ? bias[col] : 0.0f;
#pragma unroll
      for (int r = 0; r < 8; r++) {              // C layout: row = r + 8*lh
        int row = m0 + wm * 64 + mt * 16 + r + lh * 8;
        float vv = acc[mt][nt][r] + bval;
        if constexpr (RES)  vv += res[(size_t)row * N + col];
        if constexpr (RELU) vv = fmaxf(vv, 0.0f);
        if constexpr (sizeof(TC) == 2)
          ((u16*)C)[(size_t)row * N + col] = f2bf(vv);
        else
          ((float*)C)[(size_t)row * N + col] = vv;
      }
    }
}

// ---------------------------------------------------------------------------
// Flash attention: grid (S/64, H, B), 128 threads (4 waves).
// Wave w owns 16 query rows; streams 64-key tiles (16 WMMAs/tile); online
// softmax with 16-lane xor reductions; scale = 1/sqrt(d_model) = 1/32.
// K tile fetched by TDM; V tile transposed into LDS for the P@V B-operand.
// ---------------------------------------------------------------------------
__global__ __launch_bounds__(128) void attn_kernel(
    const u16* __restrict__ q, const u16* __restrict__ k,
    const u16* __restrict__ v, float* __restrict__ o)
{
  constexpr int LDK = 72, LDV = 72, LDP = 72;
  __shared__ u16 sK[64 * LDK];   // [key][e]    (B layout for Q@K^T)
  __shared__ u16 sV[64 * LDV];   // [e][key]    (B layout for P@V)
  __shared__ u16 sP[64 * LDP];   // per-wave 16x64 P scratch
  const int tid  = threadIdx.x;
  const int lane = tid & 31, wave = tid >> 5;
  const int lh = lane >> 4, lidx = lane & 15;
  const int qb = blockIdx.x, h = blockIdx.y, b = blockIdx.z;
  const int kb = lh * 8, kb2 = lh * 16;
  const unsigned sKoff = (unsigned)(size_t)(void*)sK;

  // Preload Q fragments (16x64 -> two 16x32 A frags) straight from global
  size_t qrow = (size_t)(b * Sc + qb * 64 + wave * 16 + lidx) * Dm + h * DHc;
  FragBF qf0, qf1;
  qf0.h[0] = *(const bf16x8*)(q + qrow + kb);
  qf0.h[1] = *(const bf16x8*)(q + qrow + kb + 16);
  qf1.h[0] = *(const bf16x8*)(q + qrow + 32 + kb);
  qf1.h[1] = *(const bf16x8*)(q + qrow + 32 + kb + 16);

  float mrun[8], lrun[8];
#pragma unroll
  for (int r = 0; r < 8; r++) { mrun[r] = -1e30f; lrun[r] = 0.0f; }
  f32x8 oacc[4] = {};

  for (int kt = 0; kt < Sc / 64; kt++) {
#if HAVE_TDM
    if (tid < 32) {                               // K tile: 64 keys x 64 dims
      tdm_load_2d(k + (size_t)(b * Sc + kt * 64) * Dm + h * DHc,
                  sKoff, Dm, 64, 64, 4, 3);       // 128B rows + 16B pad
      wait_tensor0();
    }
#else
#pragma unroll
    for (int i = 0; i < 4; i++) {
      int chunk = tid + i * 128;
      int r = chunk >> 3, c8 = (chunk & 7) * 8;
      *(uint4*)&sK[r * LDK + c8] =
          *(const uint4*)(k + (size_t)(b * Sc + kt * 64 + r) * Dm + h * DHc + c8);
    }
#endif
    {                                             // V tile: transposed scatter
      int t = tid >> 1, e0 = (tid & 1) * 32;
      const u16* src = v + (size_t)(b * Sc + kt * 64 + t) * Dm + h * DHc + e0;
#pragma unroll
      for (int j = 0; j < 32; j++) sV[(e0 + j) * LDV + t] = src[j];
    }
    __syncthreads();

    // scores: four 16x16 key tiles, K=64 as 2 WMMAs each
    f32x8 sc[4] = {};
#pragma unroll
    for (int ct = 0; ct < 4; ct++) {
      FragBF kf;
      kf.h[0] = *(const bf16x8*)&sK[(ct * 16 + lidx) * LDK + kb];
      kf.h[1] = *(const bf16x8*)&sK[(ct * 16 + lidx) * LDK + kb + 16];
      sc[ct] = wmma_bf16(qf0.v, kf.v, sc[ct]);
      kf.h[0] = *(const bf16x8*)&sK[(ct * 16 + lidx) * LDK + 32 + kb];
      kf.h[1] = *(const bf16x8*)&sK[(ct * 16 + lidx) * LDK + 32 + kb + 16];
      sc[ct] = wmma_bf16(qf1.v, kf.v, sc[ct]);
    }

    const float scl = 0.03125f;                   // 1/sqrt(d_model)
#pragma unroll
    for (int r = 0; r < 8; r++) {                 // row = r + 8*lh
      float a0 = sc[0][r] * scl, a1 = sc[1][r] * scl;
      float a2 = sc[2][r] * scl, a3 = sc[3][r] * scl;
      float mx = fmaxf(fmaxf(a0, a1), fmaxf(a2, a3));
      mx = fmaxf(mx, __shfl_xor(mx, 1));
      mx = fmaxf(mx, __shfl_xor(mx, 2));
      mx = fmaxf(mx, __shfl_xor(mx, 4));
      mx = fmaxf(mx, __shfl_xor(mx, 8));          // row max within 16-lane half
      float mnew = fmaxf(mrun[r], mx);
      float p0 = __expf(a0 - mnew), p1 = __expf(a1 - mnew);
      float p2 = __expf(a2 - mnew), p3 = __expf(a3 - mnew);
      float ls = (p0 + p1) + (p2 + p3);
      ls += __shfl_xor(ls, 1); ls += __shfl_xor(ls, 2);
      ls += __shfl_xor(ls, 4); ls += __shfl_xor(ls, 8);
      float corr = __expf(mrun[r] - mnew);
      lrun[r] = lrun[r] * corr + ls;
      mrun[r] = mnew;
#pragma unroll
      for (int et = 0; et < 4; et++) oacc[et][r] *= corr;
      int prow = wave * 16 + r + lh * 8;
      sP[prow * LDP + lidx]      = f2bf(p0);
      sP[prow * LDP + 16 + lidx] = f2bf(p1);
      sP[prow * LDP + 32 + lidx] = f2bf(p2);
      sP[prow * LDP + 48 + lidx] = f2bf(p3);
    }

    // P (16x64) @ V (64x64): per-wave LDS relayout -> two A fragments
    FragBF pf0, pf1;
    pf0.h[0] = *(const bf16x8*)&sP[(wave * 16 + lidx) * LDP + kb];
    pf0.h[1] = *(const bf16x8*)&sP[(wave * 16 + lidx) * LDP + kb + 16];
    pf1.h[0] = *(const bf16x8*)&sP[(wave * 16 + lidx) * LDP + 32 + kb];
    pf1.h[1] = *(const bf16x8*)&sP[(wave * 16 + lidx) * LDP + 32 + kb + 16];
#pragma unroll
    for (int et = 0; et < 4; et++) {
      FragBF vf;
      vf.h[0] = *(const bf16x8*)&sV[(et * 16 + lidx) * LDV + kb];
      vf.h[1] = *(const bf16x8*)&sV[(et * 16 + lidx) * LDV + kb + 16];
      oacc[et] = wmma_bf16(pf0.v, vf.v, oacc[et]);
      vf.h[0] = *(const bf16x8*)&sV[(et * 16 + lidx) * LDV + 32 + kb];
      vf.h[1] = *(const bf16x8*)&sV[(et * 16 + lidx) * LDV + 32 + kb + 16];
      oacc[et] = wmma_bf16(pf1.v, vf.v, oacc[et]);
    }
    __syncthreads();
  }

#pragma unroll
  for (int et = 0; et < 4; et++)
#pragma unroll
    for (int r = 0; r < 8; r++) {
      int row = b * Sc + qb * 64 + wave * 16 + r + lh * 8;
      int col = h * DHc + et * 16 + lidx;
      o[(size_t)row * Dm + col] = oacc[et][r] / lrun[r];
    }
}

// ---------------------------------------------------------------------------
// Torch-style LayerNorm: (x-mean)/(std+eps)*g + b, std with ddof=1, eps=1e-6
// ---------------------------------------------------------------------------
__global__ __launch_bounds__(256) void ln_kernel(
    const float* __restrict__ in, const float* __restrict__ g,
    const float* __restrict__ be, float* __restrict__ out)
{
  __shared__ float red[256];
  __shared__ float sMean, sInv;
  const size_t base = (size_t)blockIdx.x * Dm;
  float s = 0;
  for (int i = threadIdx.x; i < Dm; i += 256) s += in[base + i];
  red[threadIdx.x] = s; __syncthreads();
  for (int st = 128; st > 0; st >>= 1) {
    if (threadIdx.x < st) red[threadIdx.x] += red[threadIdx.x + st];
    __syncthreads();
  }
  if (threadIdx.x == 0) sMean = red[0] * (1.0f / Dm);
  __syncthreads();
  const float mean = sMean;
  float vs = 0;
  for (int i = threadIdx.x; i < Dm; i += 256) {
    float d = in[base + i] - mean; vs += d * d;
  }
  red[threadIdx.x] = vs; __syncthreads();
  for (int st = 128; st > 0; st >>= 1) {
    if (threadIdx.x < st) red[threadIdx.x] += red[threadIdx.x + st];
    __syncthreads();
  }
  if (threadIdx.x == 0)
    sInv = 1.0f / (sqrtf(red[0] * (1.0f / (Dm - 1))) + 1e-6f);
  __syncthreads();
  const float inv = sInv;
  for (int i = threadIdx.x; i < Dm; i += 256)
    out[base + i] = g[i] * (in[base + i] - mean) * inv + be[i];
}

// Mean pool over sequence: [B,S,D] -> [B,D]
__global__ __launch_bounds__(256) void pool_kernel(
    const float* __restrict__ in, float* __restrict__ outp)
{
  int idx = blockIdx.x * 256 + threadIdx.x;   // Bc*Dm
  int b = idx >> 10, d = idx & (Dm - 1);
  float s = 0;
  for (int t = 0; t < Sc; t++)
    s += in[(size_t)(b * Sc + t) * Dm + d];
  outp[idx] = s * (1.0f / Sc);
}

// Final head: out[b,j] = pooled[b,:] . Wlast[:,j] + blast[j]; grid = B*2
__global__ __launch_bounds__(256) void head_kernel(
    const float* __restrict__ pooled, const float* __restrict__ Wl,
    const float* __restrict__ bl, float* __restrict__ out)
{
  __shared__ float red[256];
  int b = blockIdx.x >> 1, j = blockIdx.x & 1;
  float s = 0;
  for (int d = threadIdx.x; d < Dm; d += 256)
    s += pooled[b * Dm + d] * Wl[d * 2 + j];
  red[threadIdx.x] = s; __syncthreads();
  for (int st = 128; st > 0; st >>= 1) {
    if (threadIdx.x < st) red[threadIdx.x] += red[threadIdx.x + st];
    __syncthreads();
  }
  if (threadIdx.x == 0) out[b * 2 + j] = red[0] + bl[j];
}

// ---------------------------------------------------------------------------
extern "C" void kernel_launch(void* const* d_in, const int* in_sizes, int n_in,
                              void* d_out, int out_size, void* d_ws, size_t ws_size,
                              hipStream_t stream)
{
  (void)in_sizes; (void)n_in; (void)out_size; (void)ws_size;
  const int*   tokens = (const int*)  d_in[0];
  const float* emb    = (const float*)d_in[1];
  const float* Wq     = (const float*)d_in[2];
  const float* Wk     = (const float*)d_in[3];
  const float* Wv     = (const float*)d_in[4];
  const float* Wo     = (const float*)d_in[5];
  const float* bo     = (const float*)d_in[6];
  const float* ln1a   = (const float*)d_in[7];
  const float* ln1b   = (const float*)d_in[8];
  const float* W1     = (const float*)d_in[9];
  const float* b1     = (const float*)d_in[10];
  const float* W2     = (const float*)d_in[11];
  const float* b2     = (const float*)d_in[12];
  const float* ln2a   = (const float*)d_in[13];
  const float* ln2b   = (const float*)d_in[14];
  const float* lnfa   = (const float*)d_in[15];
  const float* lnfb   = (const float*)d_in[16];
  const float* Wlast  = (const float*)d_in[17];
  const float* blast  = (const float*)d_in[18];
  float* out = (float*)d_out;

  char* ws = (char*)d_ws;
  size_t off = 0;
  auto take = [&](size_t bytes) -> void* {
    void* p = ws + off;
    off += (bytes + 255) & ~(size_t)255;
    return p;
  };
  const size_t MD = (size_t)Mrows * Dm;                       // 8M elems
  u16*   Bq  = (u16*)  take((size_t)Lc * Dm * Dm * 2);        // 12MB each
  u16*   Bk  = (u16*)  take((size_t)Lc * Dm * Dm * 2);
  u16*   Bv  = (u16*)  take((size_t)Lc * Dm * Dm * 2);
  u16*   BoT = (u16*)  take((size_t)Lc * Dm * Dm * 2);
  u16*   B1T = (u16*)  take((size_t)Lc * Dm * DFFc * 2);      // 48MB
  u16*   B2T = (u16*)  take((size_t)Lc * Dm * DFFc * 2);      // 48MB
  float* x   = (float*)take(MD * 4);
  float* x2  = (float*)take(MD * 4);
  float* tmp = (float*)take(MD * 4);
  float* oB  = (float*)take(MD * 4);
  u16*   qB  = (u16*)  take(MD * 2);
  u16*   kB  = (u16*)  take(MD * 2);
  u16*   vB  = (u16*)  take(MD * 2);
  u16*   hB  = (u16*)  take((size_t)Mrows * DFFc * 2);        // 64MB
  float* pooled = (float*)take((size_t)Bc * Dm * 4);

  // --- weight prep: transpose + f32->bf16 ---
  transpose_bf16_kernel<<<dim3(256, 1, Lc * Hh), 256, 0, stream>>>(
      Wq, Bq, Dm, DHc, (long long)Dm * DHc, (long long)Dm * DHc);
  transpose_bf16_kernel<<<dim3(256, 1, Lc * Hh), 256, 0, stream>>>(
      Wk, Bk, Dm, DHc, (long long)Dm * DHc, (long long)Dm * DHc);
  transpose_bf16_kernel<<<dim3(256, 1, Lc * Hh), 256, 0, stream>>>(
      Wv, Bv, Dm, DHc, (long long)Dm * DHc, (long long)Dm * DHc);
  transpose_bf16_kernel<<<dim3(4096, 1, Lc), 256, 0, stream>>>(
      Wo, BoT, Dm, Dm, (long long)Dm * Dm, (long long)Dm * Dm);
  transpose_bf16_kernel<<<dim3(16384, 1, Lc), 256, 0, stream>>>(
      W1, B1T, Dm, DFFc, (long long)Dm * DFFc, (long long)Dm * DFFc);
  transpose_bf16_kernel<<<dim3(16384, 1, Lc), 256, 0, stream>>>(
      W2, B2T, DFFc, Dm, (long long)Dm * DFFc, (long long)Dm * DFFc);

  embed_kernel<<<(Mrows * Dm) / 256, 256, 0, stream>>>(tokens, emb, x);

  dim3 gD (Dm   / 128, Mrows / 128);   // (8, 64)
  dim3 gFF(DFFc / 128, Mrows / 128);   // (32, 64)
  for (int l = 0; l < Lc; l++) {
    const u16* bq = Bq  + (size_t)l * Dm * Dm;
    const u16* bk = Bk  + (size_t)l * Dm * Dm;
    const u16* bv = Bv  + (size_t)l * Dm * Dm;
    const u16* bw = BoT + (size_t)l * Dm * Dm;
    const u16* w1 = B1T + (size_t)l * Dm * DFFc;
    const u16* w2 = B2T + (size_t)l * Dm * DFFc;

    gemm_bf16_kernel<float, u16, false, false, false><<<gD, 256, 0, stream>>>(
        x, bq, nullptr, nullptr, qB, Mrows, Dm, Dm);
    gemm_bf16_kernel<float, u16, false, false, false><<<gD, 256, 0, stream>>>(
        x, bk, nullptr, nullptr, kB, Mrows, Dm, Dm);
    gemm_bf16_kernel<float, u16, false, false, false><<<gD, 256, 0, stream>>>(
        x, bv, nullptr, nullptr, vB, Mrows, Dm, Dm);

    attn_kernel<<<dim3(Sc / 64, Hh, Bc), 128, 0, stream>>>(qB, kB, vB, oB);

    // o @ Wo + bo + residual(x) -> tmp ; LN1 -> x2
    gemm_bf16_kernel<float, float, true, true, false><<<gD, 256, 0, stream>>>(
        oB, bw, bo + (size_t)l * Dm, x, tmp, Mrows, Dm, Dm);
    ln_kernel<<<Mrows, 256, 0, stream>>>(tmp, ln1a + (size_t)l * Dm,
                                         ln1b + (size_t)l * Dm, x2);
    // relu(x2 @ W1 + b1) -> hB (bf16)
    gemm_bf16_kernel<float, u16, true, false, true><<<gFF, 256, 0, stream>>>(
        x2, w1, b1 + (size_t)l * DFFc, nullptr, hB, Mrows, DFFc, Dm);
    // hB @ W2 + b2 + residual(x2) -> tmp ; LN2 -> x
    gemm_bf16_kernel<u16, float, true, true, false><<<gD, 256, 0, stream>>>(
        hB, w2, b2 + (size_t)l * Dm, x2, tmp, Mrows, Dm, DFFc);
    ln_kernel<<<Mrows, 256, 0, stream>>>(tmp, ln2a + (size_t)l * Dm,
                                         ln2b + (size_t)l * Dm, x);
  }

  ln_kernel<<<Mrows, 256, 0, stream>>>(x, lnfa, lnfb, tmp);
  pool_kernel<<<(Bc * Dm) / 256, 256, 0, stream>>>(tmp, pooled);
  head_kernel<<<Bc * 2, 256, 0, stream>>>(pooled, Wlast, blast, out);
}
